// GNN_82325933130116
// MI455X (gfx1250) — matrix operand order repaired and checked by reading, not compiled, not run on hardware
//
#include <hip/hip_runtime.h>
#include <hip/hip_bf16.h>
#include <math.h>

typedef _Float16 v16h __attribute__((ext_vector_type(16)));
typedef float    v8f  __attribute__((ext_vector_type(8)));

union F16Frag { v16h v; float4 f4[2]; };

#define N_NODES 100000
#define N_EDGES 800000

static __device__ __forceinline__ void atomic_add_f32(float* p, float v) {
  __hip_atomic_fetch_add(p, v, __ATOMIC_RELAXED, __HIP_MEMORY_SCOPE_AGENT);
}

// ---------------- utility ----------------
__global__ void fill_f32(float* p, int n, float val) {
  int id = blockIdx.x * blockDim.x + threadIdx.x;
  if (id < n) p[id] = val;
}

__global__ void degree_kernel(const int* __restrict__ dst, float* __restrict__ deg) {
  int e = blockIdx.x * blockDim.x + threadIdx.x;
  if (e < N_EDGES) atomic_add_f32(&deg[dst[e]], 1.0f);
}

// ---------------- GCN layer 1 linear: [N,16] @ [16,128] ----------------
__global__ void gcn1_linear(const float* __restrict__ x, const float* __restrict__ W1,
                            float* __restrict__ out) {
  __shared__ float xs[32];
  int tid = threadIdx.x;
  if (tid < 32) {
    int nl = tid >> 4, k = tid & 15;
    xs[tid] = x[(blockIdx.x * 2 + nl) * 16 + k];
  }
  __syncthreads();
  int li = tid >> 7, c = tid & 127;
  int node = blockIdx.x * 2 + li;
  float acc = 0.f;
#pragma unroll
  for (int k = 0; k < 16; ++k)
    acc = fmaf(xs[li * 16 + k], W1[k * 128 + c], acc);
  out[(size_t)node * 128 + c] = acc;
}

// ---------------- scatter-sum (edges + self loops), 4 channels/thread ----------------
template <int C>
__global__ void scatter_kernel(const float* __restrict__ h, float* __restrict__ agg,
                               const int* __restrict__ src, const int* __restrict__ dst) {
  const int C4 = C / 4;
  int id = blockIdx.x * blockDim.x + threadIdx.x;
  int e = id / C4, c = (id % C4) * 4;
  int s, d;
  if (e < N_EDGES) { s = src[e]; d = dst[e]; }
  else             { s = d = e - N_EDGES; }
  float4 v = *(const float4*)(h + (size_t)s * C + c);
  float* a = agg + (size_t)d * C + c;
  atomic_add_f32(a + 0, v.x);
  atomic_add_f32(a + 1, v.y);
  atomic_add_f32(a + 2, v.z);
  atomic_add_f32(a + 3, v.w);
}

// ---------------- batchnorm: per-channel sum / sumsq ----------------
template <int C>
__global__ void bn_stats(const float* __restrict__ agg, const float* __restrict__ deg,
                         const float* __restrict__ bias,
                         float* __restrict__ gsum, float* __restrict__ gsumsq) {
  int c = threadIdx.x;
  int i0 = blockIdx.x * 250;
  float b = bias[c];
  float s = 0.f, ss = 0.f;
  for (int r = 0; r < 250; ++r) {
    int i = i0 + r;
    float v = agg[(size_t)i * C + c] * (1.0f / deg[i]) + b;
    s += v;
    ss = fmaf(v, v, ss);
  }
  atomic_add_f32(&gsum[c], s);
  atomic_add_f32(&gsumsq[c], ss);
}

__global__ void bn_finalize(const float* __restrict__ gsum, const float* __restrict__ gsumsq,
                            const float* __restrict__ gamma, const float* __restrict__ beta,
                            float* __restrict__ ss, int C) {
  int c = threadIdx.x;
  float mu = gsum[c] * (1.0f / N_NODES);
  float var = gsumsq[c] * (1.0f / N_NODES) - mu * mu;
  float rs = rsqrtf(var + 1e-5f);
  float sc = gamma[c] * rs;
  ss[c] = sc;
  ss[C + c] = beta[c] - mu * sc;
}

// ---------------- batchnorm apply + leaky relu (+ f16 copy, optional f32 out) ----------------
template <int C>
__global__ void bn_apply(const float* __restrict__ agg, const float* __restrict__ deg,
                         const float* __restrict__ bias, const float* __restrict__ ss,
                         _Float16* __restrict__ out16, float* __restrict__ out32) {
  int id = blockIdx.x * blockDim.x + threadIdx.x;
  int i = id / C, c = id % C;
  float v = agg[(size_t)id] * (1.0f / deg[i]) + bias[c];
  float y = fmaf(ss[c], v, ss[C + c]);
  y = (y > 0.f) ? y : 0.01f * y;
  if (out32) out32[(size_t)id] = y;
  out16[(size_t)id] = (_Float16)y;
}

// ---------------- pack row-major fp32 W[K,Ncols] into WMMA B-fragment f16 layout ----------------
// packed[((kb*NT + nt)*32 + lane)*16 + t]:
//   lane<16 : column n = nt*16+lane,       K = kb*32 + (t&7) + 16*(t>>3)       (t=0..15)
//   lane>=16: column n = nt*16+lane-16,    K = kb*32 + 8 + (t&7) + 16*(t>>3)
__global__ void pack_b(const float* __restrict__ W, _Float16* __restrict__ packed,
                       int Ncols, int NT) {
  int id = blockIdx.x * blockDim.x + threadIdx.x;
  int t = id & 15;
  int lane = (id >> 4) & 31;
  int kbnt = id >> 9;
  int kb = kbnt / NT, nt = kbnt % NT;
  int k, n;
  if (lane < 16) { k = kb * 32 + (t & 7) + ((t >> 3) << 4);     n = nt * 16 + lane; }
  else           { k = kb * 32 + 8 + (t & 7) + ((t >> 3) << 4); n = nt * 16 + lane - 16; }
  packed[id] = (_Float16)W[k * Ncols + n];
}

// ---------------- GCN layer 2 linear via WMMA: h1f16[N,128] @ W2p -> [N,256] f32 ----------------
__global__ void gemm_h1w2(const _Float16* __restrict__ A16, const _Float16* __restrict__ Bp,
                          float* __restrict__ Out) {
  int tid = threadIdx.x;
  int lane = tid & 31, w = tid >> 5;
  int m0 = blockIdx.x * 16;
  int mrow = m0 + (lane & 15);
  int koff = (lane & 16) ? 8 : 0;
  F16Frag af[4];
#pragma unroll
  for (int kb = 0; kb < 4; ++kb) {
    const _Float16* ap = A16 + (size_t)mrow * 128 + kb * 32 + koff;
    af[kb].f4[0] = *(const float4*)(ap);
    af[kb].f4[1] = *(const float4*)(ap + 16);
  }
#pragma unroll
  for (int t = 0; t < 2; ++t) {
    int nt = w + t * 8;
    v8f acc = {0.f, 0.f, 0.f, 0.f, 0.f, 0.f, 0.f, 0.f};
#pragma unroll
    for (int kb = 0; kb < 4; ++kb) {
      F16Frag bf;
      const float4* bp = (const float4*)(Bp + ((size_t)(kb * 16 + nt) * 32 + lane) * 16);
      bf.f4[0] = bp[0];
      bf.f4[1] = bp[1];
      acc = __builtin_amdgcn_wmma_f32_16x16x32_f16(false, af[kb].v, false, bf.v,
                                                   (short)0, acc, false, false);
    }
    int col = nt * 16 + (lane & 15);
    int rbase = m0 + ((lane & 16) ? 8 : 0);
#pragma unroll
    for (int v = 0; v < 8; ++v)
      Out[(size_t)(rbase + v) * 256 + col] = acc[v];
  }
}

// ---------------- edge MLP: concat(h2[src],h2[dst]) [512] -> 128 -> 64 -> 1 ----------------
// layer 1 AND layer 2 on WMMA; async global->LDS gather of edge features.
__global__ void edge_mlp(const _Float16* __restrict__ h2,
                         const int* __restrict__ src, const int* __restrict__ dst,
                         const _Float16* __restrict__ Wp1p, const float* __restrict__ bp1,
                         const _Float16* __restrict__ Wp2p, const float* __restrict__ bp2,
                         const float* __restrict__ Wp3, const float* __restrict__ bp3,
                         float* __restrict__ pred) {
  __shared__ __align__(16) _Float16 ef[16 * 520];   // 16 edges x 512 features (+pad)
  __shared__ __align__(16) _Float16 p1h[16 * 136];  // layer-1 activations, f16 A-layout
  __shared__ __align__(16) float p2s[16 * 68];      // layer-2 activations
  int tid = threadIdx.x;
  int e0 = blockIdx.x * 16;

  // async gather: each thread moves 32 f16 (64B) of one edge's feature vector
  // directly global -> LDS (ASYNCcnt-tracked, no VGPR round-trip)
  {
    int el = tid >> 4, seg = tid & 15;
    int e = e0 + el;
    int f = seg * 32;
    int node = (f < 256) ? src[e] : dst[e];
    if (f >= 256) f -= 256;
    unsigned ldso = (unsigned)(size_t)(&ef[el * 520 + seg * 32]);
    unsigned long long ga = (unsigned long long)(size_t)(h2 + (size_t)node * 256 + f);
    asm volatile(
        "global_load_async_to_lds_b128 %0, %1, off\n\t"
        "global_load_async_to_lds_b128 %0, %1, off offset:16\n\t"
        "global_load_async_to_lds_b128 %0, %1, off offset:32\n\t"
        "global_load_async_to_lds_b128 %0, %1, off offset:48"
        :
        : "v"(ldso), "v"(ga)
        : "memory");
    asm volatile("s_wait_asynccnt 0x0" ::: "memory");
  }
  __syncthreads();

  // layer 1: 512 -> 128 with WMMA f16. 8 waves, each owns one 16-col tile.
  int lane = tid & 31, w = tid >> 5;
  int er = lane & 15;
  int koff = (lane & 16) ? 8 : 0;
  int rb = (lane & 16) ? 8 : 0;
  {
    v8f acc = {0.f, 0.f, 0.f, 0.f, 0.f, 0.f, 0.f, 0.f};
#pragma unroll
    for (int kb = 0; kb < 16; ++kb) {
      F16Frag af, bf;
      const _Float16* ap = &ef[er * 520 + kb * 32 + koff];
      af.f4[0] = *(const float4*)(ap);
      af.f4[1] = *(const float4*)(ap + 16);
      const float4* bp = (const float4*)(Wp1p + ((size_t)(kb * 8 + w) * 32 + lane) * 16);
      bf.f4[0] = bp[0];
      bf.f4[1] = bp[1];
      acc = __builtin_amdgcn_wmma_f32_16x16x32_f16(false, af.v, false, bf.v,
                                                   (short)0, acc, false, false);
    }
    int col = w * 16 + er;
    float bb = bp1[col];
#pragma unroll
    for (int v = 0; v < 8; ++v) {
      float y = acc[v] + bb;
      p1h[(rb + v) * 136 + col] = (_Float16)((y > 0.f) ? y : 0.f);
    }
  }
  __syncthreads();

  // layer 2: 128 -> 64 with WMMA f16 on waves 0..3 (K=128 -> 4 k-blocks, N=64 -> 4 tiles)
  if (w < 4) {
    v8f acc = {0.f, 0.f, 0.f, 0.f, 0.f, 0.f, 0.f, 0.f};
#pragma unroll
    for (int kb = 0; kb < 4; ++kb) {
      F16Frag af, bf;
      const _Float16* ap = &p1h[er * 136 + kb * 32 + koff];
      af.f4[0] = *(const float4*)(ap);
      af.f4[1] = *(const float4*)(ap + 16);
      const float4* bp = (const float4*)(Wp2p + ((size_t)(kb * 4 + w) * 32 + lane) * 16);
      bf.f4[0] = bp[0];
      bf.f4[1] = bp[1];
      acc = __builtin_amdgcn_wmma_f32_16x16x32_f16(false, af.v, false, bf.v,
                                                   (short)0, acc, false, false);
    }
    int col = w * 16 + er;
    float bb = bp2[col];
#pragma unroll
    for (int v = 0; v < 8; ++v)
      p2s[(rb + v) * 68 + col] = fmaxf(acc[v] + bb, 0.f);
  }
  __syncthreads();

  // layer 3: 64 -> 1 + sigmoid
  if (tid < 16) {
    int e = tid;
    float s = bp3[0];
#pragma unroll 8
    for (int k = 0; k < 64; ++k) s = fmaf(p2s[e * 68 + k], Wp3[k], s);
    pred[e0 + e] = 1.0f / (1.0f + expf(-s));
  }
}

// ---------------- launch ----------------
extern "C" void kernel_launch(void* const* d_in, const int* in_sizes, int n_in,
                              void* d_out, int out_size, void* d_ws, size_t ws_size,
                              hipStream_t stream) {
  const float* x   = (const float*)d_in[0];
  const int*   ei  = (const int*)d_in[1];
  const int*   src = ei;
  const int*   dst = ei + N_EDGES;
  const float* W1  = (const float*)d_in[2];
  const float* b1  = (const float*)d_in[3];
  const float* g1  = (const float*)d_in[4];
  const float* be1 = (const float*)d_in[5];
  const float* W2  = (const float*)d_in[6];
  const float* b2  = (const float*)d_in[7];
  const float* g2  = (const float*)d_in[8];
  const float* be2 = (const float*)d_in[9];
  const float* Wp1 = (const float*)d_in[10];
  const float* bp1 = (const float*)d_in[11];
  const float* Wp2 = (const float*)d_in[12];
  const float* bp2 = (const float*)d_in[13];
  const float* Wp3 = (const float*)d_in[14];
  const float* bp3 = (const float*)d_in[15];
  float* out = (float*)d_out;

  char* ws = (char*)d_ws;
  float*    bufA   = (float*)(ws + 0);              // 102.4 MB : hlin1 then hlin2
  float*    bufB   = (float*)(ws + 102400000);      // 102.4 MB : agg1 then agg2
  _Float16* h1f16  = (_Float16*)(ws + 204800000);   // 25.6 MB
  _Float16* h2f16  = (_Float16*)(ws + 230400000);   // 51.2 MB
  float*    deg    = (float*)(ws + 281600000);      // 0.4 MB
  float*    gsum   = (float*)(ws + 282000000);      // 1 KB
  float*    gsumsq = gsum + 256;                    // 1 KB
  float*    ssbuf  = gsum + 512;                    // 2 KB
  _Float16* w2p    = (_Float16*)(ws + 282004096);   // 64 KB
  _Float16* wp1p   = (_Float16*)(ws + 282069632);   // 128 KB
  _Float16* wp2p   = (_Float16*)(ws + 282200704);   // 16 KB

  // degrees (self-loop => init 1)
  fill_f32<<<(N_NODES + 255) / 256, 256, 0, stream>>>(deg, N_NODES, 1.0f);
  degree_kernel<<<(N_EDGES + 255) / 256, 256, 0, stream>>>(dst, deg);

  // pack weights to WMMA B layout (independent of activations)
  pack_b<<<32768 / 256, 256, 0, stream>>>(W2, w2p, 256, 16);  // K=128,N=256
  pack_b<<<65536 / 256, 256, 0, stream>>>(Wp1, wp1p, 128, 8); // K=512,N=128
  pack_b<<<8192 / 256, 256, 0, stream>>>(Wp2, wp2p, 64, 4);   // K=128,N=64

  // ---- GCN layer 1 ----
  gcn1_linear<<<N_NODES / 2, 256, 0, stream>>>(x, W1, bufA);
  fill_f32<<<(N_NODES * 128 + 255) / 256, 256, 0, stream>>>(bufB, N_NODES * 128, 0.f);
  scatter_kernel<128><<<(900000 * 32) / 256, 256, 0, stream>>>(bufA, bufB, src, dst);
  fill_f32<<<2, 256, 0, stream>>>(gsum, 512, 0.f);
  bn_stats<128><<<400, 128, 0, stream>>>(bufB, deg, b1, gsum, gsumsq);
  bn_finalize<<<1, 128, 0, stream>>>(gsum, gsumsq, g1, be1, ssbuf, 128);
  bn_apply<128><<<(N_NODES * 128) / 256, 256, 0, stream>>>(bufB, deg, b1, ssbuf, h1f16,
                                                           (float*)nullptr);

  // ---- GCN layer 2 ----
  gemm_h1w2<<<N_NODES / 16, 256, 0, stream>>>(h1f16, w2p, bufA);
  fill_f32<<<(N_NODES * 256 + 255) / 256, 256, 0, stream>>>(bufB, N_NODES * 256, 0.f);
  scatter_kernel<256><<<(900000 * 64) / 256, 256, 0, stream>>>(bufA, bufB, src, dst);
  fill_f32<<<2, 256, 0, stream>>>(gsum, 512, 0.f);
  bn_stats<256><<<400, 256, 0, stream>>>(bufB, deg, b2, gsum, gsumsq);
  bn_finalize<<<1, 256, 0, stream>>>(gsum, gsumsq, g2, be2, ssbuf, 256);
  bn_apply<256><<<(N_NODES * 256) / 256, 256, 0, stream>>>(bufB, deg, b2, ssbuf, h2f16, out);

  // ---- edge predictor MLP ----
  edge_mlp<<<N_EDGES / 16, 256, 0, stream>>>(h2f16, src, dst, wp1p, bp1, wp2p, bp2, Wp3, bp3,
                                             out + 25600000);
}
